// GraphWithSocialAttentionEmbedding_81930796138743
// MI455X (gfx1250) — compile-verified, matrix operand order checked by reading, not run on hardware
//
#include <hip/hip_runtime.h>
#include <stdint.h>
#include <stddef.h>

// ---------------- types ----------------
typedef __bf16 bf16_t;
typedef __attribute__((ext_vector_type(16))) __bf16 v16bf;
typedef __attribute__((ext_vector_type(8)))  float  v8f;
typedef unsigned int u32x4 __attribute__((ext_vector_type(4)));

#define K_NBR  10
#define D_DIM  128
#define T_DIM  128
#define QD_DIM 256          // D + T
#define KD_DIM 384          // 2D + T
#define B_SZ   2048
#define CHUNK  4096         // hop-1 rows per chunk
#define NCHUNK 5            // 20480 / 4096

__device__ __forceinline__ unsigned short f2bf(float f) {
  unsigned int u = __float_as_uint(f);
  unsigned int r = (u + 0x7FFFu + ((u >> 16) & 1u)) >> 16;   // RNE truncate
  return (unsigned short)r;
}
__device__ __forceinline__ float bf2f(unsigned short h) {
  return __uint_as_float(((unsigned int)h) << 16);
}

// ---------------- weight fp32 -> bf16 ----------------
__global__ void cvt_f32_bf16(const float* __restrict__ in,
                             unsigned short* __restrict__ out, int n) {
  int i = blockIdx.x * 256 + threadIdx.x;
  if (i < n) out[i] = f2bf(in[i]);
}

// ---------------- build query-side inputs + mask ----------------
// q_in[i] = [ base(src_id) (128) || cos(time_b) (128) ]  (bf16)
// src_feat[i] = base(src_id)  (f32, for MergeLayer concat)
// maskbuf[i]: bits 0..9 = neighbor==0 (masked); bit31 = row invalid
__global__ void build_q(const float* __restrict__ nf, const float* __restrict__ mem,
                        const int* __restrict__ src_ids, const int* __restrict__ mask_nbrs,
                        const float* __restrict__ tb, int r0,
                        unsigned short* __restrict__ q_in, float* __restrict__ src_feat,
                        int* __restrict__ maskbuf) {
  int i  = blockIdx.x;
  int gi = r0 + i;
  int t  = threadIdx.x;                       // 0..127
  int n2 = src_ids[gi];
  float f = nf[(size_t)n2 * D_DIM + t] + mem[(size_t)n2 * D_DIM + t];
  src_feat[(size_t)i * D_DIM + t] = f;
  q_in[(size_t)i * QD_DIM + t]          = f2bf(f);
  q_in[(size_t)i * QD_DIM + D_DIM + t]  = f2bf(__cosf(tb[t]));   // time_encode(0)
  if (t == 0) {
    int bits = 0;
    for (int kk = 0; kk < K_NBR; ++kk)
      bits |= (mask_nbrs[(size_t)gi * K_NBR + kk] == 0) ? (1 << kk) : 0;
    if (bits == 0x3FF) bits = (0x3FF & ~(1 << (K_NBR - 1))) | (int)0x80000000;
    maskbuf[i] = bits;
  }
}

// ---------------- build key-side inputs ----------------
// k_in[row] = [ nbr emb (128) || edge feat (128) || cos(dt*w+b) (128) ]  (bf16)
// mode 0: nbr emb = node_features+memory gathered; mode 1: nbr emb = emb1 row.
__global__ void build_k(const float* __restrict__ nf, const float* __restrict__ mem,
                        const float* __restrict__ emb1, const float* __restrict__ ef,
                        const int* __restrict__ nbrs, const int* __restrict__ eidx,
                        const float* __restrict__ etimes, const float* __restrict__ ts,
                        const float* __restrict__ tw, const float* __restrict__ tb,
                        int r0, int tsdiv, int mode,
                        unsigned short* __restrict__ k_in) {
  int ik = blockIdx.x;                 // local k-row within chunk
  int gk = r0 * K_NBR + ik;            // global k-row
  int t  = threadIdx.x;                // 0..127
  float nv;
  if (mode == 0) {
    int nb = nbrs[gk];
    nv = nf[(size_t)nb * D_DIM + t] + mem[(size_t)nb * D_DIM + t];
  } else {
    nv = emb1[(size_t)gk * D_DIM + t];
  }
  int   e  = eidx[gk];
  float dt = ts[gk / tsdiv] - etimes[gk];
  unsigned short* kr = k_in + (size_t)ik * KD_DIM;
  kr[t]             = f2bf(nv);
  kr[D_DIM + t]     = f2bf(ef[(size_t)e * D_DIM + t]);
  kr[2 * D_DIM + t] = f2bf(__cosf(dt * tw[t] + tb[t]));
}

// ---------------- WMMA GEMM:  out[M,N] = A[M,Kd](bf16) * W[N,Kd]^T(bf16) + bias ----------------
// 1 wave -> 16x64 tile (A frag reused across 4 WMMAs), software-pipelined (double-buffered
// fragments so loads for step k+1 are in flight while WMMAs for step k issue).
// grid = (M/64, N/64), block = 128 (4 waves). flags: bit0 = bf16 output, bit1 = relu
__global__ __launch_bounds__(128) void gemm_bf16(
    const unsigned short* __restrict__ Au, const unsigned short* __restrict__ Wu,
    const float* __restrict__ bias, void* __restrict__ out,
    int N, int Kd, int flags) {
  const bf16_t* A = (const bf16_t*)Au;
  const bf16_t* W = (const bf16_t*)Wu;
  int wave = threadIdx.x >> 5;
  int lane = threadIdx.x & 31;
  int m0 = (blockIdx.x * 4 + wave) * 16;
  int n0 = blockIdx.y * 64;
  int mr = lane & 15;
  int lo = (lane < 16) ? 0 : 8;        // 16-bit operand K-striping per half-wave

  const bf16_t* arow = A + (size_t)(m0 + mr) * Kd;
  const bf16_t* wr0  = W + (size_t)(n0 + mr) * Kd;

  union U { v16bf v; u32x4 u[2]; };
  v8f acc[4] = {};

  // prologue: load k-block 0
  U a0, b0[4];
  a0.u[0] = *(const u32x4*)(arow + lo);
  a0.u[1] = *(const u32x4*)(arow + 16 + lo);
#pragma unroll
  for (int s = 0; s < 4; ++s) {
    const bf16_t* wr = wr0 + (size_t)(s * 16) * Kd;
    b0[s].u[0] = *(const u32x4*)(wr + lo);
    b0[s].u[1] = *(const u32x4*)(wr + 16 + lo);
  }

  for (int kb = 32; kb < Kd; kb += 32) {
    // issue next-step loads first (in flight during current WMMAs)
    U a1, b1[4];
    a1.u[0] = *(const u32x4*)(arow + kb + lo);
    a1.u[1] = *(const u32x4*)(arow + kb + 16 + lo);
#pragma unroll
    for (int s = 0; s < 4; ++s) {
      const bf16_t* wr = wr0 + (size_t)(s * 16) * Kd;
      b1[s].u[0] = *(const u32x4*)(wr + kb + lo);
      b1[s].u[1] = *(const u32x4*)(wr + kb + 16 + lo);
    }
#pragma unroll
    for (int s = 0; s < 4; ++s)
      acc[s] = __builtin_amdgcn_wmma_f32_16x16x32_bf16(
          false, a0.v, false, b0[s].v, (short)0, acc[s], false, false);
    a0 = a1;
#pragma unroll
    for (int s = 0; s < 4; ++s) b0[s] = b1[s];
  }
  // epilogue k-step
#pragma unroll
  for (int s = 0; s < 4; ++s)
    acc[s] = __builtin_amdgcn_wmma_f32_16x16x32_bf16(
        false, a0.v, false, b0[s].v, (short)0, acc[s], false, false);

  int ms = m0 + ((lane & 16) ? 8 : 0);
#pragma unroll
  for (int s = 0; s < 4; ++s) {
    int n = n0 + s * 16 + mr;
    float bb = bias ? bias[n] : 0.f;
#pragma unroll
    for (int j = 0; j < 8; ++j) {
      float val = acc[s][j] + bb;
      if (flags & 2) val = fmaxf(val, 0.f);
      size_t idx = (size_t)(ms + j) * N + n;
      if (flags & 1) ((unsigned short*)out)[idx] = f2bf(val);
      else           ((float*)out)[idx] = val;
    }
  }
}

// ---------------- masked multi-head attention (H=2, K=10, hd=128) ----------------
// block = 256 threads per row. q f32 [N,256]; k,v bf16 [N*K,256]; out bf16 [N,256].
// scores: 8 lanes cooperate per (head, neighbor) pair, reduced via __shfl_xor (wave32).
__global__ __launch_bounds__(256) void attention(
    const float* __restrict__ q, const unsigned short* __restrict__ k,
    const unsigned short* __restrict__ v, const int* __restrict__ maskbuf,
    unsigned short* __restrict__ out) {
  int row = blockIdx.x;
  int t   = threadIdx.x;
  __shared__ float qs[QD_DIM];
  __shared__ float sc[32];
  __shared__ float aw[32];
  qs[t] = q[(size_t)row * QD_DIM + t];
  __syncthreads();
  int mb = maskbuf[row];
  if (t < 8 * 2 * K_NBR) {             // 160 threads, 8 per pair
    int p = t >> 3;                    // pair id 0..19
    int l = t & 7;
    int h = p / K_NBR, kk = p % K_NBR;
    const unsigned short* kp = k + ((size_t)row * K_NBR + kk) * QD_DIM + h * 128;
    const float* qp = qs + h * 128;
    float s = 0.f;
#pragma unroll
    for (int d = l; d < 128; d += 8) s += qp[d] * bf2f(kp[d]);
    s += __shfl_xor(s, 1, 32);
    s += __shfl_xor(s, 2, 32);
    s += __shfl_xor(s, 4, 32);
    if (l == 0) {
      s *= 0.08838834764831845f;       // 1/sqrt(128)
      if ((mb >> kk) & 1) s = -1e9f;
      sc[p] = s;
    }
  }
  __syncthreads();
  if (t < 2) {
    float mx = -3e38f;
    for (int kk = 0; kk < K_NBR; ++kk) mx = fmaxf(mx, sc[t * K_NBR + kk]);
    float e[K_NBR]; float sum = 0.f;
    for (int kk = 0; kk < K_NBR; ++kk) { e[kk] = __expf(sc[t * K_NBR + kk] - mx); sum += e[kk]; }
    float inv = 1.f / sum;
    for (int kk = 0; kk < K_NBR; ++kk) aw[t * K_NBR + kk] = e[kk] * inv;
  }
  __syncthreads();
  int h = t >> 7;
  float o = 0.f;
#pragma unroll
  for (int kk = 0; kk < K_NBR; ++kk)
    o += aw[h * K_NBR + kk] * bf2f(v[((size_t)row * K_NBR + kk) * QD_DIM + t]);
  out[(size_t)row * QD_DIM + t] = f2bf(o);
}

// ---------------- concat [wo_out(256, zeroed if invalid) || src_feat(128)] -> bf16 ----------------
__global__ void build_fc1in(const float* __restrict__ wo_out, const float* __restrict__ src_feat,
                            const int* __restrict__ maskbuf, unsigned short* __restrict__ fc1_in) {
  int i = blockIdx.x, t = threadIdx.x;         // 384 threads
  float vl;
  if (t < QD_DIM) vl = (maskbuf[i] < 0) ? 0.f : wo_out[(size_t)i * QD_DIM + t];
  else            vl = src_feat[(size_t)i * D_DIM + (t - QD_DIM)];
  fc1_in[(size_t)i * (QD_DIM + D_DIM) + t] = f2bf(vl);
}

// ---------------- host orchestration ----------------
extern "C" void kernel_launch(void* const* d_in, const int* in_sizes, int n_in,
                              void* d_out, int out_size, void* d_ws, size_t ws_size,
                              hipStream_t stream) {
  (void)in_sizes; (void)n_in; (void)out_size; (void)ws_size;
  const float* nf  = (const float*)d_in[0];
  const float* mem = (const float*)d_in[1];
  const float* ef  = (const float*)d_in[2];
  const float* tw  = (const float*)d_in[3];
  const float* tb  = (const float*)d_in[4];
  const float* Wq  = (const float*)d_in[5];
  const float* bq  = (const float*)d_in[6];
  const float* Wk  = (const float*)d_in[7];
  const float* bk  = (const float*)d_in[8];
  const float* Wv  = (const float*)d_in[9];
  const float* bv  = (const float*)d_in[10];
  const float* Wo  = (const float*)d_in[11];
  const float* bo  = (const float*)d_in[12];
  const float* f1w = (const float*)d_in[13];
  const float* f1b = (const float*)d_in[14];
  const float* f2w = (const float*)d_in[15];
  const float* f2b = (const float*)d_in[16];
  const int*   src_nodes = (const int*)d_in[17];
  const float* ts   = (const float*)d_in[18];
  const int*   nbr2 = (const int*)d_in[19];   // [B,K] flat  (also hop-1 source ids)
  const int*   eid2 = (const int*)d_in[20];
  const float* et2  = (const float*)d_in[21];
  const int*   nbr1 = (const int*)d_in[22];   // [B*K, K] flat
  const int*   eid1 = (const int*)d_in[23];
  const float* et1  = (const float*)d_in[24];

  // bump allocator over workspace (peak ~105 MB)
  char* ws = (char*)d_ws;
  size_t off = 0;
  auto alloc = [&](size_t bytes) -> char* {
    char* p = ws + off;
    off += (bytes + 255) & ~(size_t)255;
    return p;
  };
  unsigned short* wq_bf = (unsigned short*)alloc((size_t)2 * QD_DIM * QD_DIM * 2);
  unsigned short* wk_bf = (unsigned short*)alloc((size_t)2 * QD_DIM * KD_DIM * 2);
  unsigned short* wv_bf = (unsigned short*)alloc((size_t)2 * QD_DIM * KD_DIM * 2);
  unsigned short* wo_bf = (unsigned short*)alloc((size_t)2 * QD_DIM * QD_DIM * 2);
  unsigned short* f1_bf = (unsigned short*)alloc((size_t)2 * D_DIM * (QD_DIM + D_DIM) * 2);
  unsigned short* f2_bf = (unsigned short*)alloc((size_t)2 * D_DIM * D_DIM * 2);
  float*          emb1  = (float*)alloc((size_t)B_SZ * K_NBR * D_DIM * 4);
  unsigned short* q_in  = (unsigned short*)alloc((size_t)CHUNK * QD_DIM * 2);
  unsigned short* k_in  = (unsigned short*)alloc((size_t)CHUNK * K_NBR * KD_DIM * 2);
  float*          src_f = (float*)alloc((size_t)CHUNK * D_DIM * 4);
  float*          qbuf  = (float*)alloc((size_t)CHUNK * QD_DIM * 4);
  unsigned short* kbuf  = (unsigned short*)alloc((size_t)CHUNK * K_NBR * QD_DIM * 2);
  unsigned short* vbuf  = (unsigned short*)alloc((size_t)CHUNK * K_NBR * QD_DIM * 2);
  unsigned short* atn_o = (unsigned short*)alloc((size_t)CHUNK * QD_DIM * 2);
  float*          wo_o  = (float*)alloc((size_t)CHUNK * QD_DIM * 4);
  unsigned short* fc1_i = (unsigned short*)alloc((size_t)CHUNK * (QD_DIM + D_DIM) * 2);
  unsigned short* hbuf  = (unsigned short*)alloc((size_t)CHUNK * D_DIM * 2);
  int*            maskb = (int*)alloc((size_t)CHUNK * 4);

  auto cvt = [&](const float* in, unsigned short* outp, int n) {
    cvt_f32_bf16<<<(n + 255) / 256, 256, 0, stream>>>(in, outp, n);
  };
  cvt(Wq,  wq_bf, 2 * QD_DIM * QD_DIM);
  cvt(Wk,  wk_bf, 2 * QD_DIM * KD_DIM);
  cvt(Wv,  wv_bf, 2 * QD_DIM * KD_DIM);
  cvt(Wo,  wo_bf, 2 * QD_DIM * QD_DIM);
  cvt(f1w, f1_bf, 2 * D_DIM * (QD_DIM + D_DIM));
  cvt(f2w, f2_bf, 2 * D_DIM * D_DIM);

  // ---------------- layer 0 (hop-1), chunked ----------------
  for (int ch = 0; ch < NCHUNK; ++ch) {
    int r0 = ch * CHUNK;
    build_q<<<CHUNK, 128, 0, stream>>>(nf, mem, nbr2, nbr1, tb, r0, q_in, src_f, maskb);
    build_k<<<CHUNK * K_NBR, 128, 0, stream>>>(nf, mem, (const float*)nullptr, ef,
                                               nbr1, eid1, et1, ts, tw, tb,
                                               r0, /*tsdiv=*/K_NBR * K_NBR, /*mode=*/0, k_in);
    dim3 gq(CHUNK / 64, QD_DIM / 64);
    dim3 gkv(CHUNK * K_NBR / 64, QD_DIM / 64);
    dim3 gf1(CHUNK / 64, D_DIM / 64);
    gemm_bf16<<<gq,  128, 0, stream>>>(q_in,  wq_bf, bq, qbuf, QD_DIM, QD_DIM, 0);
    gemm_bf16<<<gkv, 128, 0, stream>>>(k_in,  wk_bf, bk, kbuf, QD_DIM, KD_DIM, 1);
    gemm_bf16<<<gkv, 128, 0, stream>>>(k_in,  wv_bf, bv, vbuf, QD_DIM, KD_DIM, 1);
    attention<<<CHUNK, 256, 0, stream>>>(qbuf, kbuf, vbuf, maskb, atn_o);
    gemm_bf16<<<gq,  128, 0, stream>>>(atn_o, wo_bf, bo, wo_o, QD_DIM, QD_DIM, 0);
    build_fc1in<<<CHUNK, 384, 0, stream>>>(wo_o, src_f, maskb, fc1_i);
    gemm_bf16<<<gf1, 128, 0, stream>>>(fc1_i, f1_bf, f1b, hbuf, D_DIM, QD_DIM + D_DIM, 1 | 2);
    gemm_bf16<<<gf1, 128, 0, stream>>>(hbuf,  f2_bf, f2b, emb1 + (size_t)r0 * D_DIM, D_DIM, D_DIM, 0);
  }

  // ---------------- top layer ----------------
  {
    build_q<<<B_SZ, 128, 0, stream>>>(nf, mem, src_nodes, nbr2, tb, 0, q_in, src_f, maskb);
    build_k<<<B_SZ * K_NBR, 128, 0, stream>>>(nf, mem, emb1, ef,
                                              nbr2, eid2, et2, ts, tw, tb,
                                              0, /*tsdiv=*/K_NBR, /*mode=*/1, k_in);
    dim3 gq(B_SZ / 64, QD_DIM / 64);
    dim3 gkv(B_SZ * K_NBR / 64, QD_DIM / 64);
    dim3 gf1(B_SZ / 64, D_DIM / 64);
    gemm_bf16<<<gq,  128, 0, stream>>>(q_in,  wq_bf + QD_DIM * QD_DIM, bq + QD_DIM, qbuf, QD_DIM, QD_DIM, 0);
    gemm_bf16<<<gkv, 128, 0, stream>>>(k_in,  wk_bf + QD_DIM * KD_DIM, bk + QD_DIM, kbuf, QD_DIM, KD_DIM, 1);
    gemm_bf16<<<gkv, 128, 0, stream>>>(k_in,  wv_bf + QD_DIM * KD_DIM, bv + QD_DIM, vbuf, QD_DIM, KD_DIM, 1);
    attention<<<B_SZ, 256, 0, stream>>>(qbuf, kbuf, vbuf, maskb, atn_o);
    gemm_bf16<<<gq,  128, 0, stream>>>(atn_o, wo_bf + QD_DIM * QD_DIM, bo + QD_DIM, wo_o, QD_DIM, QD_DIM, 0);
    build_fc1in<<<B_SZ, 384, 0, stream>>>(wo_o, src_f, maskb, fc1_i);
    gemm_bf16<<<gf1, 128, 0, stream>>>(fc1_i, f1_bf + D_DIM * (QD_DIM + D_DIM), f1b + D_DIM,
                                       hbuf, D_DIM, QD_DIM + D_DIM, 1 | 2);
    gemm_bf16<<<gf1, 128, 0, stream>>>(hbuf,  f2_bf + D_DIM * D_DIM, f2b + D_DIM,
                                       (float*)d_out, D_DIM, D_DIM, 0);
  }
}
